// FineGrainedRetriever_3470333575836
// MI455X (gfx1250) — matrix operand order repaired and checked by reading, not compiled, not run on hardware
//
#include <hip/hip_runtime.h>
#include <hip/hip_bf16.h>

typedef __attribute__((ext_vector_type(16))) _Float16 v16h;
typedef __attribute__((ext_vector_type(8)))  _Float16 v8h;
typedef __attribute__((ext_vector_type(8)))  float    v8f;

#define EMB     256
#define N_NODES 20000
#define N_EDGES 200000

// ---------------------------------------------------------------------------
// WMMA fragment helpers (CDNA5 v_wmma_f32_16x16x32_f16, wave32)
// ---------------------------------------------------------------------------
// A (16x32 f16) in this file's contiguous-k layout: lane L -> row M = L&15, hs = L>>4.
//   elements 0..15 hold K = kk + hs*16 + 0..15 (matches B)
__device__ __forceinline__ v16h load_a_frag(const _Float16* lds, int KT, int kk, int lane) {
    int m  = lane & 15;
    int hs = lane >> 4;
    const _Float16* base = lds + m * KT + kk + hs * 16;
    union { v16h v; v8h h[2]; } u;
    u.h[0] = *(const v8h*)(base);
    u.h[1] = *(const v8h*)(base + 8);
    return u.v;
}

// B (32x16 f16) in this file's contiguous-k layout: lane L -> col N = ncol + (L&15),
// elements 0..15 hold K = kk + hs*16 + 0..15 (contiguous when Wt is
// column-major: Wt[n*K + k]). One 32-byte global load.
__device__ __forceinline__ v16h load_b_frag(const _Float16* __restrict__ Wt,
                                            int K, int ncol, int kk, int lane) {
    int n  = ncol + (lane & 15);
    int kb = kk + ((lane >> 4) << 4);
    return *(const v16h*)(Wt + (size_t)n * K + kb);
}

// C/D layout: element r of v8f -> row M = r + 8*hs, col N = ncol + (lane&15)

// ---------------------------------------------------------------------------
// Prep: transpose-convert f32 weights [K][Nc] -> f16 column-major [Nc][K]
// ---------------------------------------------------------------------------
__global__ void k_transpose_f16(const float* __restrict__ W, _Float16* __restrict__ Wt,
                                int K, int Nc) {
    int idx = blockIdx.x * blockDim.x + threadIdx.x;
    int total = K * Nc;
    if (idx >= total) return;
    int k = idx / Nc;
    int n = idx - k * Nc;
    Wt[(size_t)n * K + k] = (_Float16)W[(size_t)k * Nc + n];
}

__global__ void k_zero_f32(float* __restrict__ p, int n) {
    int idx = blockIdx.x * blockDim.x + threadIdx.x;
    if (idx < n) p[idx] = 0.0f;
}

// ---------------------------------------------------------------------------
// rev MLP: ea_rev = ReLU(edge_attr @ Wr1 + br1) @ Wr2 + br2   (fused 2 GEMMs)
// M=32 per block. Loop nest: kk outer -> A frags loaded once, reused by both
// column tiles (halves LDS traffic per WMMA).
// ---------------------------------------------------------------------------
__global__ void __launch_bounds__(256)
k_rev_mlp(const float* __restrict__ edge_attr,
          const _Float16* __restrict__ Wt1, const float* __restrict__ b1,
          const _Float16* __restrict__ Wt2, const float* __restrict__ b2,
          float* __restrict__ ea_rev) {
    __shared__ _Float16 ldsA[32 * EMB];   // 16 KB
    __shared__ _Float16 ldsH[32 * EMB];   // 16 KB
    const int tid  = threadIdx.x;
    const int lane = tid & 31;
    const int wave = tid >> 5;
    const int hs   = lane >> 4;
    const long e0  = (long)blockIdx.x * 32;

    // stage input tile (f32 -> f16)
    for (int i = tid; i < 32 * EMB; i += 256) {
        int r = i >> 8, k = i & 255;
        ldsA[r * EMB + k] = (_Float16)edge_attr[(e0 + r) * EMB + k];
    }
    __syncthreads();

    // GEMM 1 + bias + ReLU -> ldsH (f16)
    {
        v8f acc[2][2] = {{{}, {}}, {{}, {}}};
        for (int kk = 0; kk < EMB; kk += 32) {
            v16h a0 = load_a_frag(ldsA,            EMB, kk, lane);
            v16h a1 = load_a_frag(ldsA + 16 * EMB, EMB, kk, lane);
            #pragma unroll
            for (int ci = 0; ci < 2; ++ci) {
                v16h b = load_b_frag(Wt1, EMB, (wave + 8 * ci) * 16, kk, lane);
                acc[0][ci] = __builtin_amdgcn_wmma_f32_16x16x32_f16(false, a0, false, b, (short)0, acc[0][ci], false, false);
                acc[1][ci] = __builtin_amdgcn_wmma_f32_16x16x32_f16(false, a1, false, b, (short)0, acc[1][ci], false, false);
            }
        }
        #pragma unroll
        for (int ci = 0; ci < 2; ++ci) {
            int col = (wave + 8 * ci) * 16 + (lane & 15);
            float bias = b1[col];
            #pragma unroll
            for (int r = 0; r < 8; ++r) {
                int row = r + hs * 8;
                ldsH[row * EMB + col]        = (_Float16)fmaxf(acc[0][ci][r] + bias, 0.0f);
                ldsH[(row + 16) * EMB + col] = (_Float16)fmaxf(acc[1][ci][r] + bias, 0.0f);
            }
        }
    }
    __syncthreads();

    // GEMM 2 + bias -> global f32
    {
        v8f acc[2][2] = {{{}, {}}, {{}, {}}};
        for (int kk = 0; kk < EMB; kk += 32) {
            v16h a0 = load_a_frag(ldsH,            EMB, kk, lane);
            v16h a1 = load_a_frag(ldsH + 16 * EMB, EMB, kk, lane);
            #pragma unroll
            for (int ci = 0; ci < 2; ++ci) {
                v16h b = load_b_frag(Wt2, EMB, (wave + 8 * ci) * 16, kk, lane);
                acc[0][ci] = __builtin_amdgcn_wmma_f32_16x16x32_f16(false, a0, false, b, (short)0, acc[0][ci], false, false);
                acc[1][ci] = __builtin_amdgcn_wmma_f32_16x16x32_f16(false, a1, false, b, (short)0, acc[1][ci], false, false);
            }
        }
        #pragma unroll
        for (int ci = 0; ci < 2; ++ci) {
            int col = (wave + 8 * ci) * 16 + (lane & 15);
            float bias = b2[col];
            #pragma unroll
            for (int r = 0; r < 8; ++r) {
                int row = r + hs * 8;
                ea_rev[(e0 + row) * EMB + col]      = acc[0][ci][r] + bias;
                ea_rev[(e0 + row + 16) * EMB + col] = acc[1][ci][r] + bias;
            }
        }
    }
}

// ---------------------------------------------------------------------------
// Aggregate: s[dst_f] += x[src_f] + ea_f ; cnt[dst_f] += 1
// One thread per (edge, 4-channel group): float4 loads + 4 f32 atomics.
// ---------------------------------------------------------------------------
__global__ void k_aggregate(const float* __restrict__ xin,
                            const float* __restrict__ edge_attr,
                            const float* __restrict__ ea_rev,
                            const int* __restrict__ src,
                            const int* __restrict__ dst,
                            float* __restrict__ s, float* __restrict__ cnt) {
    long idx = (long)blockIdx.x * blockDim.x + threadIdx.x;
    const long total = 2L * N_EDGES * (EMB / 4);
    if (idx >= total) return;
    long e = idx >> 6;              // EMB/4 = 64 groups per edge
    int  g = (int)(idx & 63) * 4;
    int s_node, d_node;
    const float* ea_base;
    long erow;
    if (e < N_EDGES) {
        s_node = src[e]; d_node = dst[e];
        ea_base = edge_attr; erow = e;
    } else {
        long ee = e - N_EDGES;
        s_node = dst[ee]; d_node = src[ee];
        ea_base = ea_rev; erow = ee;
    }
    float4 ea = *(const float4*)(ea_base + erow * EMB + g);
    float4 xv = *(const float4*)(xin + (size_t)s_node * EMB + g);
    float* sd = s + (size_t)d_node * EMB + g;
    atomicAdd(sd + 0, xv.x + ea.x);
    atomicAdd(sd + 1, xv.y + ea.y);
    atomicAdd(sd + 2, xv.z + ea.z);
    atomicAdd(sd + 3, xv.w + ea.w);
    if (g == 0) atomicAdd(&cnt[d_node], 1.0f);
}

// ---------------------------------------------------------------------------
// SAGE layer output: h = ReLU([x ; s/max(cnt,1)] @ W + b)
// M=32 per block, K=512. kk-outer loop nest for A-frag reuse.
// ---------------------------------------------------------------------------
__global__ void __launch_bounds__(256)
k_sage(const float* __restrict__ xin,
       const float* __restrict__ s, const float* __restrict__ cnt,
       const _Float16* __restrict__ Wt, const float* __restrict__ bvec,
       float* __restrict__ hout) {
    __shared__ _Float16 ldsA[32 * 512];   // 32 KB
    const int tid  = threadIdx.x;
    const int lane = tid & 31;
    const int wave = tid >> 5;
    const int hs   = lane >> 4;
    const long n0  = (long)blockIdx.x * 32;

    for (int i = tid; i < 32 * 512; i += 256) {
        int r = i >> 9, k = i & 511;
        float v;
        if (k < EMB) {
            v = xin[(n0 + r) * EMB + k];
        } else {
            int c = k - EMB;
            float ct = cnt[n0 + r];
            v = s[(n0 + r) * EMB + c] / fmaxf(ct, 1.0f);
        }
        ldsA[r * 512 + k] = (_Float16)v;
    }
    __syncthreads();

    v8f acc[2][2] = {{{}, {}}, {{}, {}}};
    for (int kk = 0; kk < 512; kk += 32) {
        v16h a0 = load_a_frag(ldsA,            512, kk, lane);
        v16h a1 = load_a_frag(ldsA + 16 * 512, 512, kk, lane);
        #pragma unroll
        for (int ci = 0; ci < 2; ++ci) {
            const int ct = wave + 8 * ci;
            v16h b = load_b_frag(Wt, 512, ct * 16, kk, lane);
            if (kk + 32 < 512)
                __builtin_prefetch(Wt + (size_t)(ct * 16 + (lane & 15)) * 512 + kk + 32, 0, 1);
            acc[0][ci] = __builtin_amdgcn_wmma_f32_16x16x32_f16(false, a0, false, b, (short)0, acc[0][ci], false, false);
            acc[1][ci] = __builtin_amdgcn_wmma_f32_16x16x32_f16(false, a1, false, b, (short)0, acc[1][ci], false, false);
        }
    }
    #pragma unroll
    for (int ci = 0; ci < 2; ++ci) {
        int col = (wave + 8 * ci) * 16 + (lane & 15);
        float bias = bvec[col];
        #pragma unroll
        for (int r = 0; r < 8; ++r) {
            int row = r + hs * 8;
            hout[(n0 + row) * EMB + col]      = fmaxf(acc[0][ci][r] + bias, 0.0f);
            hout[(n0 + row + 16) * EMB + col] = fmaxf(acc[1][ci][r] + bias, 0.0f);
        }
    }
}

// ---------------------------------------------------------------------------
// Predictor: h_triple = [q ; h1[src];h2[src] ; ea ; h1[dst];h2[dst]]  (K=1536)
// logits = ReLU(h_triple @ Wp1 + bp1) @ Wp2 + bp2 ; logistic-noise sampling.
// M=64 per block; K staged in 6 chunks of 256 (32 KB LDS). kk-outer nest:
// 4 A frags loaded once per K-step, reused by both column tiles -> 8 WMMAs
// per (4 LDS + 2 global) fragment loads.
// ---------------------------------------------------------------------------
__device__ __forceinline__ float pcg_uniform(unsigned int e) {
    unsigned int state = e * 747796405u + 2891336453u;
    unsigned int word  = ((state >> ((state >> 28u) + 4u)) ^ state) * 277803737u;
    word = (word >> 22u) ^ word;
    float u = ((float)word + 0.5f) * 2.3283064365386963e-10f;
    return fminf(fmaxf(u, 1e-10f), 1.0f - 1e-7f);
}

__global__ void __launch_bounds__(256)
k_pred(const float* __restrict__ q,
       const float* __restrict__ edge_attr,
       const float* __restrict__ h1, const float* __restrict__ h2,
       const int* __restrict__ src, const int* __restrict__ dst,
       const _Float16* __restrict__ Wt1, const float* __restrict__ b1,
       const float* __restrict__ W2, const float* __restrict__ b2,
       float* __restrict__ logits_out, float* __restrict__ sampled_out) {
    const int KTOT = 1536, KC = 256, MT = 64;
    __shared__ _Float16 ldsA[MT * KC];    // 32 KB, re-staged per K-chunk
    __shared__ float lds_logit[MT];
    __shared__ int s_src[MT], s_dst[MT];
    const int tid  = threadIdx.x;
    const int lane = tid & 31;
    const int wave = tid >> 5;
    const int hs   = lane >> 4;
    const long e0  = (long)blockIdx.x * MT;

    if (tid < MT) {
        lds_logit[tid] = 0.0f;
        s_src[tid] = src[e0 + tid];
        s_dst[tid] = dst[e0 + tid];
    }
    __syncthreads();

    v8f acc[4][2];
    #pragma unroll
    for (int rt = 0; rt < 4; ++rt)
        #pragma unroll
        for (int ci = 0; ci < 2; ++ci)
            acc[rt][ci] = (v8f){};

    for (int chunk = 0; chunk < KTOT / KC; ++chunk) {
        // stage A chunk: gather triple features (f32 -> f16)
        const int kbase = chunk * KC;
        for (int i = tid; i < MT * KC; i += 256) {
            int r = i >> 8, k = i & (KC - 1);
            int gk = kbase + k;
            long e = e0 + r;
            float v;
            if (gk < 256) {
                v = q[e * EMB + gk];
            } else if (gk < 512) {
                v = h1[(size_t)s_src[r] * EMB + (gk - 256)];
            } else if (gk < 768) {
                v = h2[(size_t)s_src[r] * EMB + (gk - 512)];
            } else if (gk < 1024) {
                v = edge_attr[e * EMB + (gk - 768)];
            } else if (gk < 1280) {
                v = h1[(size_t)s_dst[r] * EMB + (gk - 1024)];
            } else {
                v = h2[(size_t)s_dst[r] * EMB + (gk - 1280)];
            }
            ldsA[r * KC + k] = (_Float16)v;
        }
        __syncthreads();

        for (int kk = 0; kk < KC; kk += 32) {
            v16h a[4];
            #pragma unroll
            for (int rt = 0; rt < 4; ++rt)
                a[rt] = load_a_frag(ldsA + rt * 16 * KC, KC, kk, lane);
            #pragma unroll
            for (int ci = 0; ci < 2; ++ci) {
                const int ct = wave + 8 * ci;
                v16h b = load_b_frag(Wt1, KTOT, ct * 16, kbase + kk, lane);
                __builtin_prefetch(Wt1 + (size_t)(ct * 16 + (lane & 15)) * KTOT + kbase + kk + 32, 0, 1);
                #pragma unroll
                for (int rt = 0; rt < 4; ++rt) {
                    acc[rt][ci] = __builtin_amdgcn_wmma_f32_16x16x32_f16(
                        false, a[rt], false, b, (short)0, acc[rt][ci], false, false);
                }
            }
        }
        __syncthreads();
    }

    // epilogue: bias + ReLU + dot with W_pred2 into LDS reduction
    #pragma unroll
    for (int ci = 0; ci < 2; ++ci) {
        const int ct = wave + 8 * ci;
        int col = ct * 16 + (lane & 15);
        float bias = b1[col];
        float w2 = W2[col];
        #pragma unroll
        for (int rt = 0; rt < 4; ++rt) {
            #pragma unroll
            for (int r = 0; r < 8; ++r) {
                int row = rt * 16 + r + hs * 8;
                float h = fmaxf(acc[rt][ci][r] + bias, 0.0f);
                atomicAdd(&lds_logit[row], h * w2);   // ds_add_f32
            }
        }
    }
    __syncthreads();

    if (tid < MT) {
        long e = e0 + tid;
        float logit = lds_logit[tid] + b2[0];
        logits_out[e] = logit;
        float u = pcg_uniform((unsigned int)e);
        float noise = __logf(u) - __logf(1.0f - u);
        sampled_out[e] = 1.0f / (1.0f + __expf(-(logit + noise)));
    }
}

// ---------------------------------------------------------------------------
// Host orchestration
// ---------------------------------------------------------------------------
extern "C" void kernel_launch(void* const* d_in, const int* in_sizes, int n_in,
                              void* d_out, int out_size, void* d_ws, size_t ws_size,
                              hipStream_t stream) {
    const float* x         = (const float*)d_in[0];
    const float* edge_attr = (const float*)d_in[1];
    const float* q         = (const float*)d_in[2];
    const int*   eidx      = (const int*)d_in[3];
    const float* W_rev1    = (const float*)d_in[4];
    const float* b_rev1    = (const float*)d_in[5];
    const float* W_rev2    = (const float*)d_in[6];
    const float* b_rev2    = (const float*)d_in[7];
    const float* W_sage1   = (const float*)d_in[8];
    const float* b_sage1   = (const float*)d_in[9];
    const float* W_sage2   = (const float*)d_in[10];
    const float* b_sage2   = (const float*)d_in[11];
    const float* W_pred1   = (const float*)d_in[12];
    const float* b_pred1   = (const float*)d_in[13];
    const float* W_pred2   = (const float*)d_in[14];
    const float* b_pred2   = (const float*)d_in[15];

    const int* src = eidx;
    const int* dst = eidx + N_EDGES;

    // workspace carve-out (256B aligned)
    char* ws = (char*)d_ws;
    size_t off = 0;
    auto carve = [&](size_t bytes) -> char* {
        char* p = ws + off;
        off = (off + bytes + 255) & ~(size_t)255;
        return p;
    };
    _Float16* WtR1 = (_Float16*)carve((size_t)EMB * EMB * 2);
    _Float16* WtR2 = (_Float16*)carve((size_t)EMB * EMB * 2);
    _Float16* WtS1 = (_Float16*)carve((size_t)2 * EMB * EMB * 2);
    _Float16* WtS2 = (_Float16*)carve((size_t)2 * EMB * EMB * 2);
    _Float16* WtP1 = (_Float16*)carve((size_t)6 * EMB * EMB * 2);
    float* ea_rev  = (float*)carve((size_t)N_EDGES * EMB * 4);
    float* sbuf    = (float*)carve((size_t)N_NODES * EMB * 4);
    float* cnt     = (float*)carve((size_t)N_NODES * 4);
    float* h1      = (float*)carve((size_t)N_NODES * EMB * 4);
    float* h2      = (float*)carve((size_t)N_NODES * EMB * 4);
    (void)ws_size; (void)n_in; (void)in_sizes; (void)out_size;

    float* logits_out  = (float*)d_out;
    float* sampled_out = (float*)d_out + N_EDGES;

    // 1) weight transpose/convert
    {
        int t = EMB * EMB;
        k_transpose_f16<<<(t + 255) / 256, 256, 0, stream>>>(W_rev1, WtR1, EMB, EMB);
        k_transpose_f16<<<(t + 255) / 256, 256, 0, stream>>>(W_rev2, WtR2, EMB, EMB);
        int t2 = 2 * EMB * EMB;
        k_transpose_f16<<<(t2 + 255) / 256, 256, 0, stream>>>(W_sage1, WtS1, 2 * EMB, EMB);
        k_transpose_f16<<<(t2 + 255) / 256, 256, 0, stream>>>(W_sage2, WtS2, 2 * EMB, EMB);
        int t6 = 6 * EMB * EMB;
        k_transpose_f16<<<(t6 + 255) / 256, 256, 0, stream>>>(W_pred1, WtP1, 6 * EMB, EMB);
    }

    // 2) reverse-edge MLP (fused two WMMA GEMMs), M=32 tiles
    k_rev_mlp<<<N_EDGES / 32, 256, 0, stream>>>(edge_attr, WtR1, b_rev1, WtR2, b_rev2, ea_rev);

    const long agg_threads = 2L * N_EDGES * (EMB / 4);
    const int  agg_blocks  = (int)((agg_threads + 255) / 256);

    // 3) SAGE layer 1
    k_zero_f32<<<(N_NODES * EMB + 255) / 256, 256, 0, stream>>>(sbuf, N_NODES * EMB);
    k_zero_f32<<<(N_NODES + 255) / 256, 256, 0, stream>>>(cnt, N_NODES);
    k_aggregate<<<agg_blocks, 256, 0, stream>>>(x, edge_attr, ea_rev, src, dst, sbuf, cnt);
    k_sage<<<N_NODES / 32, 256, 0, stream>>>(x, sbuf, cnt, WtS1, b_sage1, h1);

    // 4) SAGE layer 2
    k_zero_f32<<<(N_NODES * EMB + 255) / 256, 256, 0, stream>>>(sbuf, N_NODES * EMB);
    k_zero_f32<<<(N_NODES + 255) / 256, 256, 0, stream>>>(cnt, N_NODES);
    k_aggregate<<<agg_blocks, 256, 0, stream>>>(h1, edge_attr, ea_rev, src, dst, sbuf, cnt);
    k_sage<<<N_NODES / 32, 256, 0, stream>>>(h1, sbuf, cnt, WtS2, b_sage2, h2);

    // 5) predictor (dominant GEMM, K=1536, M=64 tiles) + sampling
    k_pred<<<N_EDGES / 64, 256, 0, stream>>>(q, edge_attr, h1, h2, src, dst,
                                             WtP1, b_pred1, W_pred2, b_pred2,
                                             logits_out, sampled_out);
}
